// MoEFFN_70901320122944
// MI455X (gfx1250) — matrix operand order repaired and checked by reading, compiled, not run on hardware
//
#include <hip/hip_runtime.h>
#include <hip/hip_bf16.h>

// ---------------- problem constants ----------------
#define TOK        4096      // B*S
#define DDIM       768
#define FDIM       1536
#define NEXP       8
#define KACT       2
#define BM         128
#define BN         128
#define BK         32
#define MAX_TILES  72        // sum ceil(Ne/128) <= 64 + 7, rounded up
#define MAX_SLOTS  9216      // 128*MAX_TILES = 9216 padded slots

typedef __bf16 bf16_t;
typedef __attribute__((ext_vector_type(16))) __bf16 v16bf;
typedef __attribute__((ext_vector_type(8)))  float  v8f;

union FragU { v16bf v; uint4 q[2]; };
union Pack4 { bf16_t b[4]; uint2 u; };

__device__ __forceinline__ v16bf load_frag(const bf16_t* p0, const bf16_t* p1) {
    FragU u;
    u.q[0] = *reinterpret_cast<const uint4*>(p0);
    u.q[1] = *reinterpret_cast<const uint4*>(p1);
    return u.v;
}

// ---------------- tiny setup kernels ----------------
__global__ void init_kernel(int* counts, int* cursor) {
    int i = threadIdx.x;
    if (i < NEXP) { counts[i] = 0; cursor[i] = 0; }
}

__global__ void router_kernel(const float* __restrict__ x,
                              const float* __restrict__ Wr,
                              const float* __restrict__ bias,
                              int* __restrict__ top_idx,
                              float* __restrict__ top_w,
                              int* __restrict__ counts) {
    int t = blockIdx.x * blockDim.x + threadIdx.x;
    if (t >= TOK) return;
    const float* xr = x + (size_t)t * DDIM;
    float lg[NEXP];
#pragma unroll
    for (int e = 0; e < NEXP; ++e) lg[e] = bias[e];
    for (int d = 0; d < DDIM; ++d) {
        float xv = xr[d];
#pragma unroll
        for (int e = 0; e < NEXP; ++e) lg[e] += xv * Wr[d * NEXP + e];
    }
    int i0 = 0;
#pragma unroll
    for (int e = 1; e < NEXP; ++e) if (lg[e] > lg[i0]) i0 = e;
    int i1 = (i0 == 0) ? 1 : 0;
#pragma unroll
    for (int e = 0; e < NEXP; ++e) if (e != i0 && lg[e] > lg[i1]) i1 = e;
    // softmax over the two selected logits (lg[i0] >= lg[i1])
    float w1 = __expf(lg[i1] - lg[i0]);
    float inv = 1.0f / (1.0f + w1);
    top_idx[2 * t]     = i0;  top_idx[2 * t + 1] = i1;
    top_w[2 * t]       = inv; top_w[2 * t + 1]   = w1 * inv;
    atomicAdd(&counts[i0], 1);
    atomicAdd(&counts[i1], 1);
}

__global__ void plan_kernel(const int* __restrict__ counts,
                            int* __restrict__ base,
                            int* __restrict__ tiles,
                            int* __restrict__ n_tiles,
                            int* __restrict__ slot_token) {
    if (threadIdx.x == 0) {
        int off = 0, nt = 0;
        for (int e = 0; e < NEXP; ++e) {
            base[e] = off;
            int ntile = (counts[e] + BM - 1) / BM;
            for (int i = 0; i < ntile; ++i) {
                tiles[2 * nt] = e;
                tiles[2 * nt + 1] = off + i * BM;
                ++nt;
            }
            off += ntile * BM;
        }
        base[NEXP] = off;
        *n_tiles = nt;
    }
    for (int i = threadIdx.x; i < MAX_SLOTS; i += blockDim.x) slot_token[i] = -1;
}

__global__ void gather_kernel(const int* __restrict__ top_idx,
                              const int* __restrict__ base,
                              int* __restrict__ cursor,
                              int* __restrict__ slot_token,
                              int* __restrict__ token_slot) {
    int t = blockIdx.x * blockDim.x + threadIdx.x;
    if (t >= TOK) return;
#pragma unroll
    for (int k = 0; k < KACT; ++k) {
        int e = top_idx[2 * t + k];
        int pos = atomicAdd(&cursor[e], 1);
        int slot = base[e] + pos;
        slot_token[slot] = t;
        token_slot[2 * t + k] = slot;
    }
}

// ---------------- WMMA GEMM core ----------------
// Loads a 32(K) x 128(N) f32 tile of B, splits to bf16 hi/lo, stores LDS as [n][k].
__device__ __forceinline__ void load_b_tile(const float* __restrict__ Bsrc,
                                            int k0, int n0, int N,
                                            bf16_t* __restrict__ Bh,
                                            bf16_t* __restrict__ Bl, int tid) {
#pragma unroll
    for (int it = 0; it < 4; ++it) {
        int chunk = tid + it * 256;          // 1024 float4 chunks
        int kr = chunk >> 5;                 // 32 float4 per k-row
        int nc = (chunk & 31) << 2;
        float4 v = *reinterpret_cast<const float4*>(Bsrc + (size_t)(k0 + kr) * N + n0 + nc);
        float vals[4] = {v.x, v.y, v.z, v.w};
#pragma unroll
        for (int j = 0; j < 4; ++j) {
            bf16_t h = (bf16_t)vals[j];
            bf16_t l = (bf16_t)(vals[j] - (float)h);
            Bh[(nc + j) * BK + kr] = h;
            Bl[(nc + j) * BK + kr] = l;
        }
    }
}

// C[M,N] = A[M,K] @ B[K,N] with fp32 accuracy via bf16 hi/lo split (3 WMMAs/term).
// FUSED: computes both B0 (gate) and B1 (up) and writes silu(g)*u.
// GATHER: A rows indirected through slot_token (negative -> zero row).
template <bool FUSED, bool GATHER>
__global__ __launch_bounds__(256) void gemm_kernel(
    const float* __restrict__ A,
    const float* __restrict__ B0,
    const float* __restrict__ B1,
    float* __restrict__ C,
    const int* __restrict__ slot_token,
    const int* __restrict__ tiles,
    const int* __restrict__ n_tiles,
    int K, int N, long bStride) {
    int mt = blockIdx.x;
    int e = 0, row_base;
    if (tiles) {
        if (mt >= *n_tiles) return;
        e = tiles[2 * mt];
        row_base = tiles[2 * mt + 1];
    } else {
        row_base = mt * BM;
    }
    const float* B0e = B0 + (size_t)e * bStride;
    const float* B1e = FUSED ? (B1 + (size_t)e * bStride) : nullptr;
    int n0 = blockIdx.y * BN;

    __shared__ __align__(16) bf16_t Ah[BM * BK];
    __shared__ __align__(16) bf16_t Al[BM * BK];
    __shared__ __align__(16) bf16_t B0h[BN * BK];
    __shared__ __align__(16) bf16_t B0l[BN * BK];
    __shared__ __align__(16) bf16_t B1h[FUSED ? BN * BK : 4];
    __shared__ __align__(16) bf16_t B1l[FUSED ? BN * BK : 4];
    __shared__ int rowTok[BM];

    int tid = threadIdx.x;
    int lane = tid & 31, wid = tid >> 5;
    int wm = wid & 3, wn = wid >> 2;     // wave tile: 32(M) x 64(N)
    int lr = lane & 15, half = lane >> 4;

    if (tid < BM) {
        int r = row_base + tid;
        rowTok[tid] = GATHER ? slot_token[r] : r;
    }
    __syncthreads();

    v8f acc0[2][4];
    v8f acc1[2][4];
#pragma unroll
    for (int ms = 0; ms < 2; ++ms)
#pragma unroll
        for (int ns = 0; ns < 4; ++ns) {
            acc0[ms][ns] = (v8f){0.f, 0.f, 0.f, 0.f, 0.f, 0.f, 0.f, 0.f};
            acc1[ms][ns] = (v8f){0.f, 0.f, 0.f, 0.f, 0.f, 0.f, 0.f, 0.f};
        }

    for (int k0 = 0; k0 < K; k0 += BK) {
        // ---- A tile: 128x32 f32 -> bf16 hi/lo, LDS [m][k]
#pragma unroll
        for (int it = 0; it < 4; ++it) {
            int chunk = tid + it * 256;      // 1024 float4 chunks
            int r = chunk >> 3;              // 8 float4 per row
            int kc = (chunk & 7) << 2;
            int tok = rowTok[r];
            float4 v = make_float4(0.f, 0.f, 0.f, 0.f);
            if (!GATHER || tok >= 0)
                v = *reinterpret_cast<const float4*>(A + (size_t)tok * K + k0 + kc);
            float vals[4] = {v.x, v.y, v.z, v.w};
            Pack4 ph, pl;
#pragma unroll
            for (int j = 0; j < 4; ++j) {
                bf16_t h = (bf16_t)vals[j];
                ph.b[j] = h;
                pl.b[j] = (bf16_t)(vals[j] - (float)h);
            }
            *reinterpret_cast<uint2*>(&Ah[r * BK + kc]) = ph.u;
            *reinterpret_cast<uint2*>(&Al[r * BK + kc]) = pl.u;
        }
        // ---- B tiles: [k][n] global -> [n][k] LDS, hi/lo
        load_b_tile(B0e, k0, n0, N, B0h, B0l, tid);
        if constexpr (FUSED) load_b_tile(B1e, k0, n0, N, B1h, B1l, tid);
        __syncthreads();

        // ---- WMMA compute
#pragma unroll
        for (int ms = 0; ms < 2; ++ms) {
            int r = wm * 32 + ms * 16 + lr;
            // A 16x32 bf16 layout: lane<16 holds K 0..7 & 16..23; lane>=16 K 8..15 & 24..31
            const bf16_t* pa = Ah + r * BK + half * 8;
            const bf16_t* pal = Al + r * BK + half * 8;
            v16bf a_h = load_frag(pa, pa + 16);
            v16bf a_l = load_frag(pal, pal + 16);
#pragma unroll
            for (int ns = 0; ns < 4; ++ns) {
                int n = wn * 64 + ns * 16 + lr;
                int kb = half * 8;           // B: lane<16 -> K 0..7 & 16..23, lane>=16 -> K 8..15 & 24..31
                const bf16_t* pb = B0h + n * BK + kb;
                const bf16_t* pbl = B0l + n * BK + kb;
                v16bf b_h = load_frag(pb, pb + 16);
                v16bf b_l = load_frag(pbl, pbl + 16);
                v8f c = acc0[ms][ns];
                c = __builtin_amdgcn_wmma_f32_16x16x32_bf16(false, a_h, false, b_h, (short)0, c, false, false);
                c = __builtin_amdgcn_wmma_f32_16x16x32_bf16(false, a_h, false, b_l, (short)0, c, false, false);
                c = __builtin_amdgcn_wmma_f32_16x16x32_bf16(false, a_l, false, b_h, (short)0, c, false, false);
                acc0[ms][ns] = c;
                if constexpr (FUSED) {
                    const bf16_t* qb = B1h + n * BK + kb;
                    const bf16_t* qbl = B1l + n * BK + kb;
                    v16bf u_h = load_frag(qb, qb + 16);
                    v16bf u_l = load_frag(qbl, qbl + 16);
                    v8f d = acc1[ms][ns];
                    d = __builtin_amdgcn_wmma_f32_16x16x32_bf16(false, a_h, false, u_h, (short)0, d, false, false);
                    d = __builtin_amdgcn_wmma_f32_16x16x32_bf16(false, a_h, false, u_l, (short)0, d, false, false);
                    d = __builtin_amdgcn_wmma_f32_16x16x32_bf16(false, a_l, false, u_h, (short)0, d, false, false);
                    acc1[ms][ns] = d;
                }
            }
        }
        __syncthreads();
    }

    // ---- epilogue: C/D layout: VGPR v, lanes 0-15 -> M=v, lanes 16-31 -> M=v+8
#pragma unroll
    for (int ms = 0; ms < 2; ++ms) {
#pragma unroll
        for (int ns = 0; ns < 4; ++ns) {
            int row = row_base + wm * 32 + ms * 16 + half * 8;
            int col = n0 + wn * 64 + ns * 16 + lr;
            float* cp = C + (size_t)row * N + col;
            v8f g = acc0[ms][ns];
            v8f u = acc1[ms][ns];
#pragma unroll
            for (int v = 0; v < 8; ++v) {
                float val;
                if constexpr (FUSED) {
                    float gv = g[v];
                    val = gv * (1.0f / (1.0f + __expf(-gv))) * u[v];
                } else {
                    val = g[v];
                }
                cp[(size_t)v * N] = val;
            }
        }
    }
}

__global__ void combine_kernel(const float* __restrict__ S_out,
                               const float* __restrict__ R_out,
                               const int* __restrict__ token_slot,
                               const float* __restrict__ top_w,
                               float* __restrict__ out) {
    int i = blockIdx.x * blockDim.x + threadIdx.x;
    if (i >= TOK * DDIM) return;
    int t = i / DDIM, d = i - t * DDIM;
    int s0 = token_slot[2 * t], s1 = token_slot[2 * t + 1];
    float r = top_w[2 * t] * R_out[(size_t)s0 * DDIM + d] +
              top_w[2 * t + 1] * R_out[(size_t)s1 * DDIM + d];
    out[i] = S_out[i] + r;
}

// ---------------- host launch ----------------
extern "C" void kernel_launch(void* const* d_in, const int* in_sizes, int n_in,
                              void* d_out, int out_size, void* d_ws, size_t ws_size,
                              hipStream_t stream) {
    const float* x     = (const float*)d_in[0];   // [T, D]
    const float* Wg_s  = (const float*)d_in[1];   // [1, D, F]
    const float* Wu_s  = (const float*)d_in[2];   // [1, D, F]
    const float* Wd_s  = (const float*)d_in[3];   // [1, F, D]
    const float* Wg_r  = (const float*)d_in[4];   // [E, D, F]
    const float* Wu_r  = (const float*)d_in[5];   // [E, D, F]
    const float* Wd_r  = (const float*)d_in[6];   // [E, F, D]
    const float* Wrt   = (const float*)d_in[7];   // [D, E]
    const float* bias  = (const float*)d_in[8];   // [E]
    float* out = (float*)d_out;

    char* ws = (char*)d_ws;
    size_t off = 0;
    auto alloc = [&](size_t bytes) {
        size_t o = off;
        off = (off + bytes + 255) & ~(size_t)255;
        return o;
    };
    float* S_out      = (float*)(ws + alloc((size_t)TOK * DDIM * 4));
    float* R_out      = (float*)(ws + alloc((size_t)MAX_SLOTS * DDIM * 4));
    float* H_s        = (float*)(ws + alloc((size_t)TOK * FDIM * 4));
    float* H_r        = (float*)(ws + alloc((size_t)MAX_SLOTS * FDIM * 4));
    int*   top_idx    = (int*)(ws + alloc((size_t)TOK * 2 * 4));
    float* top_w      = (float*)(ws + alloc((size_t)TOK * 2 * 4));
    int*   token_slot = (int*)(ws + alloc((size_t)TOK * 2 * 4));
    int*   slot_token = (int*)(ws + alloc((size_t)MAX_SLOTS * 4));
    int*   counts     = (int*)(ws + alloc(NEXP * 4));
    int*   cursor     = (int*)(ws + alloc(NEXP * 4));
    int*   base       = (int*)(ws + alloc((NEXP + 1) * 4));
    int*   tiles      = (int*)(ws + alloc(2 * MAX_TILES * 4));
    int*   n_tiles    = (int*)(ws + alloc(4));

    init_kernel<<<1, 32, 0, stream>>>(counts, cursor);
    router_kernel<<<(TOK + 127) / 128, 128, 0, stream>>>(x, Wrt, bias, top_idx, top_w, counts);
    plan_kernel<<<1, 256, 0, stream>>>(counts, base, tiles, n_tiles, slot_token);
    gather_kernel<<<(TOK + 255) / 256, 256, 0, stream>>>(top_idx, base, cursor, slot_token, token_slot);

    // shared expert gate+up fused -> H_s = silu(x@Wg_s) * (x@Wu_s)
    gemm_kernel<true, false><<<dim3(TOK / BM, FDIM / BN), 256, 0, stream>>>(
        x, Wg_s, Wu_s, H_s, nullptr, nullptr, nullptr, DDIM, FDIM, 0L);
    // routed experts gate+up fused (A rows gathered via slot_token)
    gemm_kernel<true, true><<<dim3(MAX_TILES, FDIM / BN), 256, 0, stream>>>(
        x, Wg_r, Wu_r, H_r, slot_token, tiles, n_tiles, DDIM, FDIM, (long)DDIM * FDIM);
    // shared down-proj: S_out = H_s @ Wd_s
    gemm_kernel<false, false><<<dim3(TOK / BM, DDIM / BN), 256, 0, stream>>>(
        H_s, Wd_s, nullptr, S_out, nullptr, nullptr, nullptr, FDIM, DDIM, 0L);
    // routed down-proj: R_out = H_r @ Wd_r[e]  (rows are slots already; expert from tile list)
    gemm_kernel<false, false><<<dim3(MAX_TILES, DDIM / BN), 256, 0, stream>>>(
        H_r, Wd_r, nullptr, R_out, nullptr, tiles, n_tiles, FDIM, DDIM, (long)FDIM * DDIM);

    combine_kernel<<<(TOK * DDIM + 255) / 256, 256, 0, stream>>>(S_out, R_out, token_slot, top_w, out);
}
